// SPF_88390426951918
// MI455X (gfx1250) — matrix operand order repaired and checked by reading, not compiled, not run on hardware
//
#include <hip/hip_runtime.h>
#include <hip/hip_bf16.h>
#include <math.h>

// ---------------- types / constants ----------------
typedef __bf16 bf16;
typedef __attribute__((ext_vector_type(16))) __bf16 v16bf;
typedef __attribute__((ext_vector_type(8)))  float  v8f;

#define DIMK   1024
#define NSEQ   4096
#define BATCH  4
#define HEADS  16
#define DHEAD  64
#define ROWS   (BATCH*NSEQ)      // 16384
#define BH     (BATCH*HEADS)     // 64
#define NBF    266
#define MPAD   288               // 266 padded to multiple of 32 (zero features)
#define KPS    296               // LDS row stride (bf16 elems) for qp tiles (n-major)
#define NST    40                // LDS row stride (bf16 elems) for kp tiles (m-major)
#define KEPS   1e-4f

#define DN_C     0.35355339059327373f   // 64^-0.25
#define RATIO_C  0.061313924562f        // 266^-0.5

// ---------------- WMMA helpers ----------------
static __device__ __forceinline__ v8f wmma_bf16(v16bf a, v16bf b, v8f c) {
  return __builtin_amdgcn_wmma_f32_16x16x32_bf16(false, a, false, b, (short)0, c, false, false);
}

union FragU { v16bf v; uint4 u[2]; };
union Half8U { uint4 u; bf16 h[8]; };

// A fragment (16xK bf16, row-major source, ld in elems).
// Lane layout: row = <row>, halves 0-7 -> k0+8*hi.., halves 8-15 -> k0+16+8*hi..
static __device__ __forceinline__ v16bf load_a_frag(const bf16* p, int ld, int row, int k0, int hi) {
  const bf16* r = p + (size_t)row * ld + k0 + 8 * hi;
  FragU f;
  f.u[0] = *(const uint4*)(r);
  f.u[1] = *(const uint4*)(r + 16);
  return f.v;
}

// B fragment (Kx16 bf16). Source stored "N-major": srcNK[col][k] contiguous in k.
// Lane layout: col = <col>, halves 0-7 -> k0+8*hi.., halves 8-15 -> k0+16+8*hi..
static __device__ __forceinline__ v16bf load_b_frag(const bf16* p, int ld, int col, int k0, int hi) {
  const bf16* r = p + (size_t)col * ld + k0 + 8 * hi;
  FragU f;
  f.u[0] = *(const uint4*)(r);
  f.u[1] = *(const uint4*)(r + 16);
  return f.v;
}

// order-preserving float<->uint map for atomic max
static __device__ __forceinline__ unsigned fmap(float f) {
  unsigned u = __float_as_uint(f);
  return (u >> 31) ? ~u : (u | 0x80000000u);
}
static __device__ __forceinline__ float funmap(unsigned u) {
  unsigned v = (u & 0x80000000u) ? (u & 0x7fffffffu) : ~u;
  return __uint_as_float(v);
}

// ---------------- conversion kernels ----------------
__global__ void k_cvt_bf16(const float* __restrict__ src, bf16* __restrict__ dst, int n) {
  int i = blockIdx.x * blockDim.x + threadIdx.x;
  if (i < n) dst[i] = (bf16)src[i];
}

__global__ void k_cvt_proj(const float* __restrict__ proj, bf16* __restrict__ dst) {
  int i = blockIdx.x * blockDim.x + threadIdx.x;   // over MPAD*DHEAD
  if (i >= MPAD * DHEAD) return;
  int m = i / DHEAD;
  dst[i] = (m < NBF) ? (bf16)proj[i] : (bf16)0.0f;
}

// ---------------- big GEMM: out = A(bf16)[R,1024] @ W(bf16)[1024,1024]^T ----------------
// Software-pipelined: fragments for k+1 are loaded (one clause) before the k WMMA chain.
// mode 0/1/2: scatter-store bf16 into qkv[mode] laid out [b,h,n,64]
// mode 3:     store f32 + bias into outf [R,1024]
__global__ __launch_bounds__(256) void k_gemm_xwT(
    const bf16* __restrict__ A, const bf16* __restrict__ W0, const bf16* __restrict__ W1,
    const bf16* __restrict__ W2, bf16* __restrict__ qkv, float* __restrict__ outf,
    const float* __restrict__ bias, int mode_base) {
  int mode = mode_base + blockIdx.z;
  const bf16* W = (mode == 1) ? W1 : (mode == 2) ? W2 : W0;
  int lane = threadIdx.x & 31, wave = threadIdx.x >> 5;
  int li = lane & 15, hi = lane >> 4;
  int wm = wave & 3, wn = wave >> 2;
  int rowBase = blockIdx.y * 128 + wm * 32;
  int colBase = blockIdx.x * 128 + wn * 64;

  v8f acc[2][4];
#pragma unroll
  for (int i = 0; i < 2; ++i)
#pragma unroll
    for (int j = 0; j < 4; ++j) acc[i][j] = (v8f){};

  v16bf a0 = load_a_frag(A, DIMK, rowBase + li, 0, hi);
  v16bf a1 = load_a_frag(A, DIMK, rowBase + 16 + li, 0, hi);
  v16bf bf0 = load_b_frag(W, DIMK, colBase + 0 * 16 + li, 0, hi);
  v16bf bf1 = load_b_frag(W, DIMK, colBase + 1 * 16 + li, 0, hi);
  v16bf bf2 = load_b_frag(W, DIMK, colBase + 2 * 16 + li, 0, hi);
  v16bf bf3 = load_b_frag(W, DIMK, colBase + 3 * 16 + li, 0, hi);

  for (int k0 = 0; k0 < DIMK; k0 += 32) {
    int kn = k0 + 32;
    v16bf na0, na1, nb0, nb1, nb2, nb3;
    if (kn < DIMK) {
      na0 = load_a_frag(A, DIMK, rowBase + li, kn, hi);
      na1 = load_a_frag(A, DIMK, rowBase + 16 + li, kn, hi);
      nb0 = load_b_frag(W, DIMK, colBase + 0 * 16 + li, kn, hi);
      nb1 = load_b_frag(W, DIMK, colBase + 1 * 16 + li, kn, hi);
      nb2 = load_b_frag(W, DIMK, colBase + 2 * 16 + li, kn, hi);
      nb3 = load_b_frag(W, DIMK, colBase + 3 * 16 + li, kn, hi);
      __builtin_prefetch(A + (size_t)(rowBase + li) * DIMK + kn + 32, 0, 1);
    }
    acc[0][0] = wmma_bf16(a0, bf0, acc[0][0]);
    acc[1][0] = wmma_bf16(a1, bf0, acc[1][0]);
    acc[0][1] = wmma_bf16(a0, bf1, acc[0][1]);
    acc[1][1] = wmma_bf16(a1, bf1, acc[1][1]);
    acc[0][2] = wmma_bf16(a0, bf2, acc[0][2]);
    acc[1][2] = wmma_bf16(a1, bf2, acc[1][2]);
    acc[0][3] = wmma_bf16(a0, bf3, acc[0][3]);
    acc[1][3] = wmma_bf16(a1, bf3, acc[1][3]);
    a0 = na0; a1 = na1; bf0 = nb0; bf1 = nb1; bf2 = nb2; bf3 = nb3;
  }

#pragma unroll
  for (int i = 0; i < 2; ++i) {
#pragma unroll
    for (int j = 0; j < 4; ++j) {
      int col = colBase + j * 16 + li;
#pragma unroll
      for (int g = 0; g < 8; ++g) {
        int r = rowBase + i * 16 + g + 8 * hi;
        float v = acc[i][j][g];
        if (mode < 3) {
          int b = r >> 12, n = r & 4095;
          int h = col >> 6, e = col & 63;
          qkv[(size_t)mode * ROWS * DIMK +
              (((size_t)(b * HEADS + h) * NSEQ + n) * DHEAD) + e] = (bf16)v;
        } else {
          outf[(size_t)r * DIMK + col] = v + bias[col];
        }
      }
    }
  }
}

// ---------------- diag = 0.5*dn^2*sum(d^2) per row ----------------
__global__ void k_diag(const bf16* __restrict__ src, float* __restrict__ dst, int nrows) {
  int r = blockIdx.x * blockDim.x + threadIdx.x;
  if (r >= nrows) return;
  const uint4* p = (const uint4*)(src + (size_t)r * DHEAD);
  float s = 0.f;
#pragma unroll
  for (int i = 0; i < 8; ++i) {
    Half8U c; c.u = p[i];
#pragma unroll
    for (int j = 0; j < 8; ++j) { float x = (float)c.h[j]; s += x * x; }
  }
  dst[r] = s * 0.0625f;   // 0.5 * dn^2, dn^2 = 1/8
}

__global__ void k_init_gmax(unsigned* g) { *g = 0u; }

// ---------------- global max of key data_dash (stab for keys) ----------------
__global__ __launch_bounds__(256) void k_kstab(const bf16* __restrict__ kmat,
                                               const bf16* __restrict__ projKM,
                                               unsigned* __restrict__ gmax) {
  __shared__ __align__(16) bf16 pl[NBF * DHEAD];
  __shared__ float red[256];
  // stage proj into LDS via gfx1250 async global->LDS copies (ASYNCcnt)
  {
    uint32_t lbase = (uint32_t)(uintptr_t)pl;
    for (int c = threadIdx.x; c < (NBF * DHEAD) / 8; c += blockDim.x) {
      uint32_t loff = lbase + c * 16;
      uint64_t g = (uint64_t)(uintptr_t)(projKM + c * 8);
      asm volatile("global_load_async_to_lds_b128 %0, %1, off"
                   :: "v"(loff), "v"(g) : "memory");
    }
    asm volatile("s_wait_asynccnt 0x0" ::: "memory");
  }
  __syncthreads();
  int r = blockIdx.x * blockDim.x + threadIdx.x;  // 0..BH*NSEQ-1
  float kr[DHEAD];
  const uint4* p = (const uint4*)(kmat + (size_t)r * DHEAD);
#pragma unroll
  for (int i = 0; i < 8; ++i) {
    Half8U c; c.u = p[i];
#pragma unroll
    for (int j = 0; j < 8; ++j) kr[i * 8 + j] = (float)c.h[j];
  }
  float mx = -INFINITY;
  for (int m = 0; m < NBF; ++m) {
    const uint4* q4 = (const uint4*)(pl + m * DHEAD);
    float d = 0.f;
#pragma unroll
    for (int i = 0; i < 8; ++i) {
      Half8U c; c.u = q4[i];
#pragma unroll
      for (int j = 0; j < 8; ++j) d += kr[i * 8 + j] * (float)c.h[j];
    }
    mx = fmaxf(mx, d);
  }
  mx *= DN_C;  // dn>0 commutes with max
  red[threadIdx.x] = mx;
  __syncthreads();
  for (int s = 128; s > 0; s >>= 1) {
    if (threadIdx.x < s) red[threadIdx.x] = fmaxf(red[threadIdx.x], red[threadIdx.x + s]);
    __syncthreads();
  }
  if (threadIdx.x == 0) atomicMax(gmax, fmap(red[0]));
}

// ---------------- fused key path: kp never leaves LDS ----------------
// One WG (4 waves) per (b,h). kp tile stored m-major [MPAD][NST] so phase-A writes
// are ds_store_b128 and phase-B B-fragments are contiguous ds_load_b128.
__global__ __launch_bounds__(128) void k_context(
    const bf16* __restrict__ kmat, const bf16* __restrict__ vmat,
    const bf16* __restrict__ projKM, const float* __restrict__ diagK,
    const unsigned* __restrict__ gmax, bf16* __restrict__ ctxT,
    float* __restrict__ ksum) {
  __shared__ __align__(16) bf16 kpL[MPAD * NST];   // m-major: kpL[m*NST + n]
  __shared__ __align__(16) bf16 vT[DHEAD * 32];    // vT[e*32 + n]
  __shared__ float diagL[32];
  __shared__ float ksumL[MPAD];
  int bh = blockIdx.x;
  int tid = threadIdx.x;
  int lane = tid & 31, wave = tid >> 5;
  int li = lane & 15, hi = lane >> 4;
  float kstab = funmap(*gmax);
  for (int i = tid; i < MPAD; i += 128) ksumL[i] = 0.f;

  v8f acc[18];
#pragma unroll
  for (int t = 0; t < 18; ++t) acc[t] = (v8f){};

  const bf16* kbase = kmat + (size_t)bh * NSEQ * DHEAD;
  const bf16* vbase = vmat + (size_t)bh * NSEQ * DHEAD;

  for (int n0 = 0; n0 < NSEQ; n0 += 32) {
    __syncthreads();  // previous tile fully consumed
    // stage v^T [e][n] and diag for this 32-row tile
    for (int i = tid; i < 32 * DHEAD; i += 128) {
      int n = i >> 6, e = i & 63;
      vT[e * 32 + n] = vbase[(size_t)(n0 + n) * DHEAD + e];
    }
    if (tid < 32) diagL[tid] = diagK[(size_t)bh * NSEQ + n0 + tid];
    __syncthreads();

    // phase A: this wave computes 9 of 36 kp tiles (2 n-subtiles x 18 m-tiles)
#pragma unroll
    for (int t9 = 0; t9 < 9; ++t9) {
      int t = wave * 9 + t9;
      int nsub = t & 1, mt = t >> 1;
      v8f d = (v8f){};
#pragma unroll
      for (int ks = 0; ks < 2; ++ks) {
        v16bf a = load_a_frag(kbase, DHEAD, n0 + nsub * 16 + li, ks * 32, hi);
        v16bf b = load_b_frag(projKM, DHEAD, mt * 16 + li, ks * 32, hi);
        d = wmma_bf16(a, b, d);
      }
      int m = mt * 16 + li;
      Half8U w;  // 8 consecutive n values for fixed m -> one b128 LDS store
#pragma unroll
      for (int g = 0; g < 8; ++g) {
        int nl = nsub * 16 + 8 * hi + g;
        float dash = DN_C * d[g];
        float kp = (m < NBF) ? RATIO_C * (__expf(dash - diagL[nl] - kstab) + KEPS) : 0.f;
        w.h[g] = (bf16)kp;
      }
      *(uint4*)(kpL + (size_t)m * NST + nsub * 16 + 8 * hi) = w.u;
    }
    __syncthreads();

    // phase B: context^T accumulation; wave owns one 16-row e-tile
    {
      v16bf a0 = load_a_frag(vT, 32, wave * 16 + li, 0, hi);  // K = n (32)
#pragma unroll
      for (int mt = 0; mt < 18; ++mt) {
        v16bf b = load_b_frag(kpL, NST, mt * 16 + li, 0, hi); // contiguous b128 pair
        acc[mt] = wmma_bf16(a0, b, acc[mt]);
      }
    }
    // phase C: k_cumsum (contiguous reads in m-major layout)
    for (int m = tid; m < MPAD; m += 128) {
      const bf16* row = kpL + (size_t)m * NST;
      float s = 0.f;
#pragma unroll
      for (int n = 0; n < 32; ++n) s += (float)row[n];
      ksumL[m] += s;
    }
  }
  __syncthreads();

#pragma unroll
  for (int mt = 0; mt < 18; ++mt) {
    int m = mt * 16 + li;
#pragma unroll
    for (int g = 0; g < 8; ++g) {
      int e = wave * 16 + g + 8 * hi;
      ctxT[((size_t)bh * DHEAD + e) * MPAD + m] = (bf16)acc[mt][g];
    }
  }
  for (int m = tid; m < MPAD; m += 128) ksum[(size_t)bh * MPAD + m] = ksumL[m];
}

// ---------------- fused query path: qp in LDS only, out -> y ----------------
__global__ __launch_bounds__(128) void k_qout(
    const bf16* __restrict__ qmat, const bf16* __restrict__ projKM,
    const float* __restrict__ diagQ, const bf16* __restrict__ ctxT,
    const float* __restrict__ ksum, bf16* __restrict__ y) {
  __shared__ float ksumL[MPAD];
  __shared__ __align__(16) bf16 qpL[4 * 16 * KPS];
  int bh = blockIdx.y;
  int chunk = blockIdx.x;  // 0..63, 64 rows each
  int tid = threadIdx.x;
  int lane = tid & 31, wave = tid >> 5;
  int li = lane & 15, hi = lane >> 4;
  for (int i = tid; i < MPAD; i += 128) ksumL[i] = ksum[(size_t)bh * MPAD + i];
  __syncthreads();

  int n0 = chunk * 64 + wave * 16;
  const bf16* qbase = qmat + (size_t)bh * NSEQ * DHEAD;

  // q_dash for all 18 m-tiles, held in registers
  v8f acc[18];
#pragma unroll
  for (int mt = 0; mt < 18; ++mt) {
    v8f d = (v8f){};
#pragma unroll
    for (int ks = 0; ks < 2; ++ks) {
      v16bf a = load_a_frag(qbase, DHEAD, n0 + li, ks * 32, hi);
      v16bf b = load_b_frag(projKM, DHEAD, mt * 16 + li, ks * 32, hi);
      d = wmma_bf16(a, b, d);
    }
    acc[mt] = d;
  }

  // per-row stab = max over m (cross-lane within 16-lane groups)
  float vmax[8], diag[8], dsum[8];
#pragma unroll
  for (int g = 0; g < 8; ++g) vmax[g] = -INFINITY;
#pragma unroll
  for (int mt = 0; mt < 18; ++mt) {
    int m = mt * 16 + li;
    if (m < NBF) {
#pragma unroll
      for (int g = 0; g < 8; ++g) vmax[g] = fmaxf(vmax[g], DN_C * acc[mt][g]);
    }
  }
#pragma unroll
  for (int g = 0; g < 8; ++g) {
#pragma unroll
    for (int s = 1; s < 16; s <<= 1) vmax[g] = fmaxf(vmax[g], __shfl_xor(vmax[g], s, 16));
    diag[g] = diagQ[(size_t)bh * NSEQ + n0 + g + 8 * hi];
    dsum[g] = 0.f;
  }

  // qp -> LDS (bf16) + D^-1 accumulation
  bf16* myqp = qpL + wave * 16 * KPS;
#pragma unroll
  for (int mt = 0; mt < 18; ++mt) {
    int m = mt * 16 + li;
    float km = ksumL[m];
#pragma unroll
    for (int g = 0; g < 8; ++g) {
      float qp = (m < NBF) ? RATIO_C * (__expf(DN_C * acc[mt][g] - diag[g] - vmax[g]) + KEPS) : 0.f;
      myqp[(g + 8 * hi) * KPS + m] = (bf16)qp;
      dsum[g] += qp * km;
    }
  }
  float dinv[8];
#pragma unroll
  for (int g = 0; g < 8; ++g) {
#pragma unroll
    for (int s = 1; s < 16; s <<= 1) dsum[g] += __shfl_xor(dsum[g], s, 16);
    dinv[g] = 1.f / dsum[g];
  }

  // out[16 n][64 e] = qp @ context ; context[m][e] == ctxT[e][m]
  const bf16* cbase = ctxT + (size_t)bh * DHEAD * MPAD;
  int b = bh >> 4, h = bh & 15;
#pragma unroll
  for (int et = 0; et < 4; ++et) {
    v8f o = (v8f){};
#pragma unroll
    for (int ks = 0; ks < 9; ++ks) {
      v16bf a = load_a_frag(myqp, KPS, li, ks * 32, hi);
      v16bf bb = load_b_frag(cbase, MPAD, et * 16 + li, ks * 32, hi);
      o = wmma_bf16(a, bb, o);
    }
    int e = et * 16 + li;
#pragma unroll
    for (int g = 0; g < 8; ++g) {
      int n = n0 + g + 8 * hi;
      y[((size_t)(b * NSEQ) + n) * DIMK + h * DHEAD + e] = (bf16)(o[g] * dinv[g]);
    }
  }
}

// ---------------- launcher ----------------
extern "C" void kernel_launch(void* const* d_in, const int* in_sizes, int n_in,
                              void* d_out, int out_size, void* d_ws, size_t ws_size,
                              hipStream_t stream) {
  (void)in_sizes; (void)n_in; (void)out_size; (void)ws_size;
  const float* x    = (const float*)d_in[0];
  const float* Wq   = (const float*)d_in[1];
  const float* Wk   = (const float*)d_in[2];
  const float* Wv   = (const float*)d_in[3];
  const float* Wo   = (const float*)d_in[4];
  const float* bo   = (const float*)d_in[5];
  const float* proj = (const float*)d_in[6];

  char* ws = (char*)d_ws;
  size_t off = 0;
  auto alloc = [&](size_t bytes) -> void* {
    void* p = ws + off;
    off = (off + bytes + 255) & ~(size_t)255;
    return p;
  };
  bf16*  xb     = (bf16*)alloc((size_t)ROWS * DIMK * 2);
  bf16*  Wqb    = (bf16*)alloc((size_t)DIMK * DIMK * 2);
  bf16*  Wkb    = (bf16*)alloc((size_t)DIMK * DIMK * 2);
  bf16*  Wvb    = (bf16*)alloc((size_t)DIMK * DIMK * 2);
  bf16*  Wob    = (bf16*)alloc((size_t)DIMK * DIMK * 2);
  bf16*  projKM = (bf16*)alloc((size_t)MPAD * DHEAD * 2);
  bf16*  qkv    = (bf16*)alloc((size_t)3 * ROWS * DIMK * 2);
  float* diagQ  = (float*)alloc((size_t)BH * NSEQ * 4);
  float* diagK  = (float*)alloc((size_t)BH * NSEQ * 4);
  unsigned* gmax = (unsigned*)alloc(256);
  bf16*  ctxT   = (bf16*)alloc((size_t)BH * DHEAD * MPAD * 2);
  float* ksum   = (float*)alloc((size_t)BH * MPAD * 4);
  bf16*  yb     = (bf16*)alloc((size_t)ROWS * DIMK * 2);

  bf16* qb = qkv;
  bf16* kb = qkv + (size_t)ROWS * DIMK;
  bf16* vb = qkv + (size_t)2 * ROWS * DIMK;

  // 1) conversions
  k_cvt_bf16<<<(ROWS * DIMK + 255) / 256, 256, 0, stream>>>(x, xb, ROWS * DIMK);
  k_cvt_bf16<<<(DIMK * DIMK + 255) / 256, 256, 0, stream>>>(Wq, Wqb, DIMK * DIMK);
  k_cvt_bf16<<<(DIMK * DIMK + 255) / 256, 256, 0, stream>>>(Wk, Wkb, DIMK * DIMK);
  k_cvt_bf16<<<(DIMK * DIMK + 255) / 256, 256, 0, stream>>>(Wv, Wvb, DIMK * DIMK);
  k_cvt_bf16<<<(DIMK * DIMK + 255) / 256, 256, 0, stream>>>(Wo, Wob, DIMK * DIMK);
  k_cvt_proj<<<(MPAD * DHEAD + 255) / 256, 256, 0, stream>>>(proj, projKM);

  // 2) QKV projections (one launch, z selects weight matrix)
  k_gemm_xwT<<<dim3(DIMK / 128, ROWS / 128, 3), 256, 0, stream>>>(
      xb, Wqb, Wkb, Wvb, qkv, nullptr, nullptr, 0);

  // 3) diag terms and global key stabilizer
  k_diag<<<(BH * NSEQ) / 256, 256, 0, stream>>>(qb, diagQ, BH * NSEQ);
  k_diag<<<(BH * NSEQ) / 256, 256, 0, stream>>>(kb, diagK, BH * NSEQ);
  k_init_gmax<<<1, 1, 0, stream>>>(gmax);
  k_kstab<<<(BH * NSEQ) / 256, 256, 0, stream>>>(kb, projKM, gmax);

  // 4) fused key path -> context^T + k_cumsum
  k_context<<<BH, 128, 0, stream>>>(kb, vb, projKM, diagK, gmax, ctxT, ksum);

  // 5) fused query path -> y (pre output-projection)
  k_qout<<<dim3(NSEQ / 64, BH), 128, 0, stream>>>(qb, projKM, diagQ, ctxT, ksum, yb);

  // 6) output projection + bias -> d_out (fp32)
  k_gemm_xwT<<<dim3(DIMK / 128, ROWS / 128, 1), 256, 0, stream>>>(
      yb, Wob, Wob, Wob, nullptr, (float*)d_out, bo, 3);
}
